// GGSNN_18700287607374
// MI455X (gfx1250) — matrix operand order repaired and checked by reading, not compiled, not run on hardware
//
#include <hip/hip_runtime.h>

typedef __attribute__((ext_vector_type(16))) _Float16 v16h;
typedef __attribute__((ext_vector_type(8)))  float    v8f;

#define DD 128          // feature dim == N (nodes per graph)
#define LD 136          // padded f16 LDS row stride (16B-aligned fragments, bank-conflict-free)
#define NSTEP 4
#define WMAT_ELEMS (DD * DD)

// ---------------- WMMA operand fragment loaders (CDNA5 16x16x32 f16 layouts) ----------------

// A fragment (16x32, MxK) from LDS row-major f16 matrix.
// Lane L: row = m_base + L%16 ; halves 0..7 -> K = c*8+0..7, halves 8..15 -> K = c*8+16..23 (c = L/16)
__device__ __forceinline__ v16h frag_a_lds(const _Float16* A, int m_base, int k_base) {
  const int lane = threadIdx.x & 31;
  const int r = m_base + (lane & 15);
  const int c = (lane >> 4) * 8;
  const _Float16* p = A + r * LD + k_base + c;
  union { v16h v; uint4 q[2]; } u;
  u.q[0] = *(const uint4*)(p);
  u.q[1] = *(const uint4*)(p + 16);
  return u.v;
}

// A fragment from a transposed LDS matrix T (T[k][m] == A[m][k]) — used for h @ U^T.
__device__ __forceinline__ v16h frag_a_ldsT(const _Float16* T, int m_base, int k_base) {
  const int lane = threadIdx.x & 31;
  const int m = m_base + (lane & 15);
  const int c = (lane >> 4) * 8;
  union { v16h v; _Float16 h[16]; } u;
#pragma unroll
  for (int i = 0; i < 8; ++i) u.h[i]     = T[(k_base + c + i) * LD + m];
#pragma unroll
  for (int i = 0; i < 8; ++i) u.h[8 + i] = T[(k_base + c + 16 + i) * LD + m];
  return u.v;
}

// A fragment straight from a global fp32 row-major 128x128 matrix (mask tile, L2-resident).
__device__ __forceinline__ v16h frag_a_gf32(const float* __restrict__ A, int m_base, int k_base) {
  const int lane = threadIdx.x & 31;
  const int r = m_base + (lane & 15);
  const int c = (lane >> 4) * 8;
  const float* p = A + r * DD + k_base + c;
  union { v16h v; _Float16 h[16]; } u;
  const float4 f0 = *(const float4*)(p + 0);
  const float4 f1 = *(const float4*)(p + 4);
  const float4 f2 = *(const float4*)(p + 16);
  const float4 f3 = *(const float4*)(p + 20);
  u.h[0]=(_Float16)f0.x; u.h[1]=(_Float16)f0.y; u.h[2]=(_Float16)f0.z; u.h[3]=(_Float16)f0.w;
  u.h[4]=(_Float16)f1.x; u.h[5]=(_Float16)f1.y; u.h[6]=(_Float16)f1.z; u.h[7]=(_Float16)f1.w;
  u.h[8]=(_Float16)f2.x; u.h[9]=(_Float16)f2.y; u.h[10]=(_Float16)f2.z; u.h[11]=(_Float16)f2.w;
  u.h[12]=(_Float16)f3.x; u.h[13]=(_Float16)f3.y; u.h[14]=(_Float16)f3.z; u.h[15]=(_Float16)f3.w;
  return u.v;
}

// B fragment (32x16, KxN): stored matrix row n is B column n (contiguous along K).
// Lane L: n = n_base + L%16 ; halves 0..7 -> K = c*8+0..7, halves 8..15 -> K = c*8+16..23 (c = L/16)
__device__ __forceinline__ v16h frag_b_lds(const _Float16* Bm, int n_base, int k_base) {
  const int lane = threadIdx.x & 31;
  const int nrow = n_base + (lane & 15);
  const int c = (lane >> 4) * 8;
  const _Float16* p = Bm + nrow * LD + k_base + c;
  union { v16h v; uint4 q[2]; } u;
  u.q[0] = *(const uint4*)(p);
  u.q[1] = *(const uint4*)(p + 16);
  return u.v;
}

// ---------------- GEMM helpers: one wave computes a 16x128 strip, K=128 ----------------
// B fragment is software-pipelined: next frag's ds_loads issue before the dependent WMMA.

__device__ __forceinline__ void wmma_row(v8f acc[8], const v16h a, const _Float16* B, int kb) {
  v16h bcur = frag_b_lds(B, 0, kb * 32);
#pragma unroll
  for (int nt = 0; nt < 8; ++nt) {
    const v16h buse = bcur;
    if (nt < 7) bcur = frag_b_lds(B, (nt + 1) * 16, kb * 32);
    acc[nt] = __builtin_amdgcn_wmma_f32_16x16x32_f16(false, a, false, buse,
                                                     (short)0, acc[nt], false, false);
  }
}

__device__ __forceinline__ void gemm_Alds(v8f acc[8], const _Float16* A, const _Float16* B, int Mbase) {
#pragma unroll
  for (int kb = 0; kb < 4; ++kb) { const v16h a = frag_a_lds(A, Mbase, kb * 32); wmma_row(acc, a, B, kb); }
}
__device__ __forceinline__ void gemm_AldsT(v8f acc[8], const _Float16* T, const _Float16* B, int Mbase) {
#pragma unroll
  for (int kb = 0; kb < 4; ++kb) { const v16h a = frag_a_ldsT(T, Mbase, kb * 32); wmma_row(acc, a, B, kb); }
}
__device__ __forceinline__ void gemm_Agf32(v8f acc[8], const float* __restrict__ A, const _Float16* B, int Mbase) {
#pragma unroll
  for (int kb = 0; kb < 4; ++kb) { const v16h a = frag_a_gf32(A, Mbase, kb * 32); wmma_row(acc, a, B, kb); }
}

// ---------------- Async weight staging: f16 weight cache (global) -> LDS via ASYNCcnt DMA path ----
// Each instruction copies 16B/lane (512B/wave); 8 per thread covers the 32KB matrix with LD-padding.
__device__ __forceinline__ void stage_weight_async(const _Float16* __restrict__ Wf16, _Float16* ws) {
  const int t = threadIdx.x;
  const unsigned lds0 = (unsigned)(size_t)ws;           // low 32 bits of LDS-aperture addr = LDS offset
#pragma unroll
  for (int i = 0; i < 8; ++i) {
    const int chunk = t + i * 256;                      // 0..2047 16-byte chunks
    const int r  = chunk >> 4;                          // source row (16 chunks per 256B row)
    const int cc = (chunk & 15) * 8;                    // halves within row
    const unsigned ldsaddr = lds0 + (unsigned)(r * LD + cc) * 2u;
    const unsigned goff    = (unsigned)(r * DD + cc) * 2u;
    asm volatile("global_load_async_to_lds_b128 %0, %1, %2"
                 :: "v"(ldsaddr), "v"(goff), "s"(Wf16) : "memory");
  }
}
__device__ __forceinline__ void wait_async_done() {
  asm volatile("s_wait_asynccnt 0x0" ::: "memory");
}

// Store f32 C accumulators (WMMA C layout) to an LDS f16 row-major matrix.
__device__ __forceinline__ void store_c16(_Float16* dst, int Mbase, const v8f acc[8]) {
  const int lane = threadIdx.x & 31;
  const int hi = (lane >> 4) * 8;
#pragma unroll
  for (int nt = 0; nt < 8; ++nt) {
    const int n = nt * 16 + (lane & 15);
#pragma unroll
    for (int j = 0; j < 8; ++j) dst[(Mbase + hi + j) * LD + n] = (_Float16)acc[nt][j];
  }
}

// ---------------- Pre-kernel: convert the 7 weight matrices fp32 -> f16 once into d_ws ----------
// Cache order: 0:Wp 1:Wz 2:Uz 3:Wr 4:Ur 5:Wh 6:Uh  (row-major, unpadded)
__global__ void __launch_bounds__(256)
cvt_weights_kernel(const float* __restrict__ Wp, const float* __restrict__ Wz,
                   const float* __restrict__ Uz, const float* __restrict__ Wr,
                   const float* __restrict__ Ur, const float* __restrict__ Wh,
                   const float* __restrict__ Uh, _Float16* __restrict__ dst) {
  const int m = blockIdx.y;
  const float* src;
  switch (m) {
    case 0: src = Wp; break; case 1: src = Wz; break; case 2: src = Uz; break;
    case 3: src = Wr; break; case 4: src = Ur; break; case 5: src = Wh; break;
    default: src = Uh; break;
  }
  const int i = blockIdx.x * 256 + threadIdx.x;
  dst[m * WMAT_ELEMS + i] = (_Float16)src[i];
}

// ---------------- Main kernel: one workgroup == one batch, runs all 4 steps ----------------

__global__ void __launch_bounds__(256, 1)
ggsnn_kernel(const float* __restrict__ hidden, const float* __restrict__ mask,
             const _Float16* __restrict__ wcache,
             const float* __restrict__ bp, const float* __restrict__ bz,
             const float* __restrict__ cz, const float* __restrict__ br,
             const float* __restrict__ cr, const float* __restrict__ bh,
             const float* __restrict__ ch,
             float* __restrict__ out)
{
  __shared__ __align__(16) _Float16 hT_s[DD * LD];     // h transposed (B of mask-GEMM, A^T for U-GEMMs)
  __shared__ __align__(16) _Float16 neigh_s[DD * LD];  // neigh (A for W-GEMMs)
  __shared__ __align__(16) _Float16 scr_s[DD * LD];    // agg, later r*h
  __shared__ __align__(16) _Float16 w_s[DD * LD];      // async-staged weight (B operand)

  const int b    = blockIdx.x;
  const int t    = threadIdx.x;
  const int lane = t & 31;
  const int Mbase = (t >> 5) * 16;       // wave id * 16
  const int hi   = (lane >> 4) * 8;

  const float* __restrict__ maskb = mask   + (size_t)b * DD * DD;
  const float* __restrict__ hsrc  = hidden + (size_t)b * DD * DD;

  const v8f vzero = {0.f,0.f,0.f,0.f,0.f,0.f,0.f,0.f};

  // Master h in fp32 registers, element (m,n) owned per WMMA C layout:
  // m = Mbase + hi + j, n = nt*16 + lane%16
  float hreg[8][8];
#pragma unroll
  for (int nt = 0; nt < 8; ++nt) {
    const int n = nt * 16 + (lane & 15);
#pragma unroll
    for (int j = 0; j < 8; ++j) hreg[nt][j] = hsrc[(Mbase + hi + j) * DD + n];
  }
  // f16 transposed copy for GEMM operands
  for (int i = t; i < DD * DD; i += 256) {
    const int r = i >> 7, cN = i & 127;
    hT_s[cN * LD + r] = (_Float16)hsrc[i];
  }
  __syncthreads();

  for (int step = 0; step < NSTEP; ++step) {
    // ---- agg = mask[b] @ h ----  (weight Wp streams in via async DMA meanwhile)
    stage_weight_async(wcache + 0 * WMAT_ELEMS, w_s);
    {
      v8f acc[8];
#pragma unroll
      for (int i = 0; i < 8; ++i) acc[i] = vzero;
      gemm_Agf32(acc, maskb, hT_s, Mbase);
      store_c16(scr_s, Mbase, acc);
    }
    wait_async_done();
    __syncthreads();

    // ---- neigh = agg @ Wp^T + 128*bp ----
    {
      v8f acc[8];
#pragma unroll
      for (int i = 0; i < 8; ++i) acc[i] = vzero;
      gemm_Alds(acc, scr_s, w_s, Mbase);
#pragma unroll
      for (int nt = 0; nt < 8; ++nt) {
        const int n = nt * 16 + (lane & 15);
        const float bias = 128.0f * bp[n];
#pragma unroll
        for (int j = 0; j < 8; ++j)
          neigh_s[(Mbase + hi + j) * LD + n] = (_Float16)(acc[nt][j] + bias);
      }
    }
    __syncthreads();

    // ---- z = sigmoid(neigh@Wz^T + h@Uz^T + bz + cz)  (kept packed f16x2 in registers) ----
    unsigned zpk[8][4];
    stage_weight_async(wcache + 1 * WMAT_ELEMS, w_s);   // Wz
    wait_async_done();
    __syncthreads();
    {
      v8f acc[8];
#pragma unroll
      for (int i = 0; i < 8; ++i) acc[i] = vzero;
      gemm_Alds(acc, neigh_s, w_s, Mbase);
      __syncthreads();
      stage_weight_async(wcache + 2 * WMAT_ELEMS, w_s); // Uz
      wait_async_done();
      __syncthreads();
      gemm_AldsT(acc, hT_s, w_s, Mbase);
#pragma unroll
      for (int nt = 0; nt < 8; ++nt) {
        const int n = nt * 16 + (lane & 15);
        const float bias = bz[n] + cz[n];
#pragma unroll
        for (int j = 0; j < 8; j += 2) {
          const float z0 = 1.0f / (1.0f + __expf(-(acc[nt][j]     + bias)));
          const float z1 = 1.0f / (1.0f + __expf(-(acc[nt][j + 1] + bias)));
          union { _Float16 h[2]; unsigned u; } pk;
          pk.h[0] = (_Float16)z0; pk.h[1] = (_Float16)z1;
          zpk[nt][j >> 1] = pk.u;
        }
      }
    }
    __syncthreads();

    // ---- r = sigmoid(neigh@Wr^T + h@Ur^T + br + cr); scr = r*h ----
    stage_weight_async(wcache + 3 * WMAT_ELEMS, w_s);   // Wr
    wait_async_done();
    __syncthreads();
    {
      v8f acc[8];
#pragma unroll
      for (int i = 0; i < 8; ++i) acc[i] = vzero;
      gemm_Alds(acc, neigh_s, w_s, Mbase);
      __syncthreads();
      stage_weight_async(wcache + 4 * WMAT_ELEMS, w_s); // Ur
      wait_async_done();
      __syncthreads();
      gemm_AldsT(acc, hT_s, w_s, Mbase);
#pragma unroll
      for (int nt = 0; nt < 8; ++nt) {
        const int n = nt * 16 + (lane & 15);
        const float bias = br[n] + cr[n];
#pragma unroll
        for (int j = 0; j < 8; ++j) {
          const float r = 1.0f / (1.0f + __expf(-(acc[nt][j] + bias)));
          scr_s[(Mbase + hi + j) * LD + n] = (_Float16)(r * hreg[nt][j]);
        }
      }
    }
    __syncthreads();

    // ---- h_hat = tanh(neigh@Wh^T + (r*h)@Uh^T + bh + ch); h = (1-z)h + z*h_hat ----
    stage_weight_async(wcache + 5 * WMAT_ELEMS, w_s);   // Wh
    wait_async_done();
    __syncthreads();
    {
      v8f acc[8];
#pragma unroll
      for (int i = 0; i < 8; ++i) acc[i] = vzero;
      gemm_Alds(acc, neigh_s, w_s, Mbase);
      __syncthreads();
      stage_weight_async(wcache + 6 * WMAT_ELEMS, w_s); // Uh
      wait_async_done();
      __syncthreads();
      gemm_Alds(acc, scr_s, w_s, Mbase);
#pragma unroll
      for (int nt = 0; nt < 8; ++nt) {
        const int n = nt * 16 + (lane & 15);
        const float bias = bh[n] + ch[n];
#pragma unroll
        for (int j = 0; j < 8; ++j) {
          const int m = Mbase + hi + j;
          const float e  = __expf(2.0f * (acc[nt][j] + bias));
          const float th = 1.0f - 2.0f / (e + 1.0f);        // tanh, overflow-safe
          union { unsigned u; _Float16 h[2]; } pk;
          pk.u = zpk[nt][j >> 1];
          const float z  = (float)pk.h[j & 1];
          const float hn = (1.0f - z) * hreg[nt][j] + z * th;
          hreg[nt][j] = hn;
          hT_s[n * LD + m] = (_Float16)hn;
        }
      }
    }
    __syncthreads();   // hT_s consistent before next step's mask-GEMM
  }

  // ---- write final h (fp32 master, each thread owns its elements) ----
#pragma unroll
  for (int nt = 0; nt < 8; ++nt) {
    const int n = nt * 16 + (lane & 15);
#pragma unroll
    for (int j = 0; j < 8; ++j)
      out[((size_t)b * DD + (Mbase + hi + j)) * DD + n] = hreg[nt][j];
  }
}

extern "C" void kernel_launch(void* const* d_in, const int* in_sizes, int n_in,
                              void* d_out, int out_size, void* d_ws, size_t ws_size,
                              hipStream_t stream) {
  const float* hidden = (const float*)d_in[0];
  const float* mask   = (const float*)d_in[1];
  const float* Wp = (const float*)d_in[2];  const float* bp = (const float*)d_in[3];
  const float* Wz = (const float*)d_in[4];  const float* bz = (const float*)d_in[5];
  const float* Uz = (const float*)d_in[6];  const float* cz = (const float*)d_in[7];
  const float* Wr = (const float*)d_in[8];  const float* br = (const float*)d_in[9];
  const float* Ur = (const float*)d_in[10]; const float* cr = (const float*)d_in[11];
  const float* Wh = (const float*)d_in[12]; const float* bh = (const float*)d_in[13];
  const float* Uh = (const float*)d_in[14]; const float* ch = (const float*)d_in[15];

  _Float16* wcache = (_Float16*)d_ws;      // 7 * 16384 f16 = 224KB
  cvt_weights_kernel<<<dim3(WMAT_ELEMS / 256, 7), 256, 0, stream>>>(
      Wp, Wz, Uz, Wr, Ur, Wh, Uh, wcache);

  const int B = in_sizes[1] / (DD * DD);   // mask is [B, 128, 128]
  ggsnn_kernel<<<B, 256, 0, stream>>>(hidden, mask, wcache,
                                      bp, bz, cz, br, cr, bh, ch,
                                      (float*)d_out);
}
